// InducingPointGP_2774548873987
// MI455X (gfx1250) — matrix operand order repaired and checked by reading, not compiled, not run on hardware
//
#include <hip/hip_runtime.h>

// ---------------------------------------------------------------------------
// InducingPointGP for MI455X (gfx1250, wave32, WMMA).
// Pipeline: stats -> standardize(bf16) -> K=RBF(Z,Z) [WMMA] -> blocked chol
// -> blocked trinv -> Kx=RBF(x,Z) [WMMA, writes x_new + bf16 Kx^T]
// -> v=Linv@Kx [WMMA, fused sum(v^2) via atomics] -> var_new.
// ---------------------------------------------------------------------------

typedef __attribute__((ext_vector_type(16))) __bf16 v16bf;
typedef __attribute__((ext_vector_type(8)))  __bf16 v8bf;
typedef __attribute__((ext_vector_type(8)))  float  v8f;

static constexpr int Bb = 64, Aa = 256, Dd = 256, Pp = 1024;
static constexpr int Nn = Bb * Aa;                 // 16384 atoms

// workspace byte offsets (~55 MB total)
static constexpr size_t OFF_STATS = 0;             // 4 KB: sum[256], sumsq[256], n
static constexpr size_t OFF_PARAMS = 4096;         // 4 KB: s[256], t[256], gexp[256], jitter
static constexpr size_t OFF_ZN    = 8192;          // 4 KB  |zg|^2 per inducing pt
static constexpr size_t OFF_XN    = 12288;         // 64 KB |xg|^2 per atom
static constexpr size_t OFF_ZG    = 131072;        // 512 KB bf16 Z*gexp (P x D)
static constexpr size_t OFF_XG    = 1048576;       // 8 MB  bf16 xs*gexp (N x D)
static constexpr size_t OFF_K     = 9437184;       // 4 MB  K / L in place (P x P f32)
static constexpr size_t OFF_LINV  = 13631488;      // 4 MB  Linv f32
static constexpr size_t OFF_LINVH = 17825792;      // 2 MB  Linv bf16
static constexpr size_t OFF_KX    = 20971520;      // 32 MB bf16 Kx^T (N x P)
static constexpr size_t OFF_VACC  = 54525952;      // 64 KB f32 sum(v^2) per atom

// ---------------------------------------------------------------------------
// Stage 1: masked per-feature sum / sumsq / count
// ---------------------------------------------------------------------------
__global__ void k_stats(const float* __restrict__ x, const bool* __restrict__ msk,
                        float* __restrict__ st) {
  const int t = threadIdx.x;          // feature
  const int r0 = blockIdx.x * 64;     // 256 blocks * 64 rows = 16384
  float s = 0.f, q = 0.f;
  for (int r = 0; r < 64; ++r) {
    const float m = msk[r0 + r] ? 1.f : 0.f;
    const float v = x[(size_t)(r0 + r) * Dd + t];
    s += v * m;
    q += v * v * m;
  }
  atomicAdd(&st[t], s);
  atomicAdd(&st[Dd + t], q);
  if (t == 0) {
    int c = 0;
    for (int r = 0; r < 64; ++r) c += msk[r0 + r] ? 1 : 0;
    atomicAdd(&st[2 * Dd], (float)c);
  }
}

__global__ void k_finalize(const float* __restrict__ st, const float* __restrict__ gamma,
                           const float* __restrict__ sigma, float* __restrict__ prm) {
  const int t = threadIdx.x;
  const float n    = st[2 * Dd];
  const float mean = st[t] / n;
  const float var  = (st[Dd + t] - n * mean * mean) / (n - 1.f);
  const float stdv = sqrtf(var) + 1e-5f;
  const float ge   = __expf(gamma[t]);
  const float s    = ge / stdv;               // xg = x*s + tshift  (== xs*gexp)
  prm[t]           = s;
  prm[Dd + t]      = -mean * s;
  prm[2 * Dd + t]  = ge;
  if (t == 0) prm[3 * Dd] = __expf(sigma[0]) + 1e-3f;   // jitter
}

// ---------------------------------------------------------------------------
// Stage 2: scale to bf16 + row norms
// ---------------------------------------------------------------------------
__global__ void k_prep_z(const float* __restrict__ Z, const float* __restrict__ prm,
                         __bf16* __restrict__ zg, float* __restrict__ zn) {
  const int p = blockIdx.x, t = threadIdx.x;
  const float v = Z[(size_t)p * Dd + t] * prm[2 * Dd + t];
  zg[(size_t)p * Dd + t] = (__bf16)v;
  __shared__ float red[256];
  red[t] = v * v;
  __syncthreads();
  for (int s = 128; s > 0; s >>= 1) { if (t < s) red[t] += red[t + s]; __syncthreads(); }
  if (t == 0) zn[p] = red[0];
}

__global__ void k_prep_x(const float* __restrict__ x, const float* __restrict__ prm,
                         __bf16* __restrict__ xg, float* __restrict__ xn) {
  const int i = blockIdx.x, t = threadIdx.x;
  const float v = x[(size_t)i * Dd + t] * prm[t] + prm[Dd + t];
  xg[(size_t)i * Dd + t] = (__bf16)v;
  __shared__ float red[256];
  red[t] = v * v;
  __syncthreads();
  for (int s = 128; s > 0; s >>= 1) { if (t < s) red[t] += red[t + s]; __syncthreads(); }
  if (t == 0) xn[i] = red[0];
}

// ---------------------------------------------------------------------------
// Shared WMMA GEMM core: C[128x128] = A(MxK,row) * B(NxK,row)^T, bf16 -> f32.
// 256 threads = 8 waves; wave w owns a 64x32 sub-tile (4x2 of 16x16).
// LDS staging with padded stride 40 (80 B) for conflict-free b128 lane reads.
// ---------------------------------------------------------------------------
__device__ __forceinline__ void gemm_core(const __bf16* __restrict__ A,
                                          const __bf16* __restrict__ B,
                                          int lda, int ldb, int kdim,
                                          int m0, int n0, v8f (&acc)[4][2]) {
  constexpr int LS = 40;
  __shared__ __bf16 As[128 * LS];
  __shared__ __bf16 Bs[128 * LS];
  const int tid  = threadIdx.x;
  const int wave = tid >> 5, lane = tid & 31;
  const int wr = wave >> 2, wc = wave & 3;
  const int lsel = lane & 15, hi = lane >> 4;

  #pragma unroll
  for (int mt = 0; mt < 4; ++mt)
    #pragma unroll
    for (int nt = 0; nt < 2; ++nt)
      #pragma unroll
      for (int e = 0; e < 8; ++e) acc[mt][nt][e] = 0.f;

  for (int kt = 0; kt < kdim; kt += 32) {
    #pragma unroll
    for (int c = 0; c < 2; ++c) {
      const int chunk = tid + c * 256;       // 512 16B chunks per operand
      const int row = chunk >> 2, part = chunk & 3;
      const uint4 ga = *(const uint4*)(A + (size_t)(m0 + row) * lda + kt + part * 8);
      *(uint4*)(&As[row * LS + part * 8]) = ga;
      const uint4 gb = *(const uint4*)(B + (size_t)(n0 + row) * ldb + kt + part * 8);
      *(uint4*)(&Bs[row * LS + part * 8]) = gb;
      if (kt + 32 < kdim) {                  // global_prefetch_b8 next k-slice
        __builtin_prefetch(A + (size_t)(m0 + row) * lda + kt + 32 + part * 8, 0, 3);
        __builtin_prefetch(B + (size_t)(n0 + row) * ldb + kt + 32 + part * 8, 0, 3);
      }
    }
    __syncthreads();

    v16bf fa[4], fb[2];
    #pragma unroll
    for (int mt = 0; mt < 4; ++mt) {         // A frag: K offs {0,16} / {8,24}
      const int r = wr * 64 + mt * 16 + lsel;
      const v8bf lo = *(const v8bf*)(&As[r * LS + hi * 8]);
      const v8bf hh = *(const v8bf*)(&As[r * LS + 16 + hi * 8]);
      fa[mt] = __builtin_shufflevector(lo, hh, 0,1,2,3,4,5,6,7,8,9,10,11,12,13,14,15);
    }
    #pragma unroll
    for (int nt = 0; nt < 2; ++nt) {         // B frag: K offs {0,16} / {8,24}
      const int r = wc * 32 + nt * 16 + lsel;
      const v8bf lo = *(const v8bf*)(&Bs[r * LS + hi * 8]);
      const v8bf hh = *(const v8bf*)(&Bs[r * LS + 16 + hi * 8]);
      fb[nt] = __builtin_shufflevector(lo, hh, 0,1,2,3,4,5,6,7,8,9,10,11,12,13,14,15);
    }
    #pragma unroll
    for (int mt = 0; mt < 4; ++mt)
      #pragma unroll
      for (int nt = 0; nt < 2; ++nt)
        acc[mt][nt] = __builtin_amdgcn_wmma_f32_16x16x32_bf16(
            false, fa[mt], false, fb[nt], (short)0, acc[mt][nt], false, false);
    __syncthreads();
  }
}

// C/D layout: col = n0 + wc*32 + nt*16 + (lane&15); row = m0 + wr*64 + mt*16 + 8*(lane>=16) + e
#define EPILOG_IDX                                              \
  const int tid = threadIdx.x, wave = tid >> 5, lane = tid & 31; \
  const int wr = wave >> 2, wc = wave & 3;                       \
  const int lsel = lane & 15, hi = lane >> 4;

// K = RBF(Z,Z) + jitter*I
__global__ void k_gemm_zz(const __bf16* __restrict__ zg, const float* __restrict__ zn,
                          const float* __restrict__ prm, float* __restrict__ Km) {
  v8f acc[4][2];
  const int m0 = blockIdx.x * 128, n0 = blockIdx.y * 128;
  gemm_core(zg, zg, Dd, Dd, Dd, m0, n0, acc);
  const float jit = prm[3 * Dd];
  EPILOG_IDX
  #pragma unroll
  for (int mt = 0; mt < 4; ++mt)
    #pragma unroll
    for (int nt = 0; nt < 2; ++nt) {
      const int col  = n0 + wc * 32 + nt * 16 + lsel;
      const int rowb = m0 + wr * 64 + mt * 16 + hi * 8;
      #pragma unroll
      for (int e = 0; e < 8; ++e) {
        const int row = rowb + e;
        const float d = fabsf(-2.f * acc[mt][nt][e] + zn[row] + zn[col]);
        float kv = __expf(-d);
        if (row == col) kv += jit;
        Km[(size_t)row * Pp + col] = kv;
      }
    }
}

// Kx^T = RBF(x,Z): writes x_new (f32, masked) and Kx^T (bf16) for the v-GEMM
__global__ void k_gemm_xz(const __bf16* __restrict__ xg, const __bf16* __restrict__ zg,
                          const float* __restrict__ xn, const float* __restrict__ zn,
                          const bool* __restrict__ msk, __bf16* __restrict__ Kx,
                          float* __restrict__ xnew) {
  v8f acc[4][2];
  const int m0 = blockIdx.x * 128, n0 = blockIdx.y * 128;   // m: atoms, n: inducing
  gemm_core(xg, zg, Dd, Dd, Dd, m0, n0, acc);
  EPILOG_IDX
  #pragma unroll
  for (int mt = 0; mt < 4; ++mt)
    #pragma unroll
    for (int nt = 0; nt < 2; ++nt) {
      const int col  = n0 + wc * 32 + nt * 16 + lsel;
      const int rowb = m0 + wr * 64 + mt * 16 + hi * 8;
      #pragma unroll
      for (int e = 0; e < 8; ++e) {
        const int row = rowb + e;
        const float d  = fabsf(-2.f * acc[mt][nt][e] + xn[row] + zn[col]);
        const float kv = __expf(-d);
        Kx[(size_t)row * Pp + col]   = (__bf16)kv;
        xnew[(size_t)row * Pp + col] = msk[row] ? kv : 0.f;
      }
    }
}

// v = Linv @ Kx : fused column sum(v^2) via f32 atomics (v never materialized)
__global__ void k_gemm_v(const __bf16* __restrict__ Wh, const __bf16* __restrict__ Kx,
                         float* __restrict__ vacc) {
  v8f acc[4][2];
  const int m0 = blockIdx.x * 128, n0 = blockIdx.y * 128;   // m: P rows, n: atoms
  gemm_core(Wh, Kx, Pp, Pp, Pp, m0, n0, acc);
  EPILOG_IDX
  (void)wr; (void)hi;
  #pragma unroll
  for (int mt = 0; mt < 4; ++mt)
    #pragma unroll
    for (int nt = 0; nt < 2; ++nt) {
      const int col = n0 + wc * 32 + nt * 16 + lsel;
      float s = 0.f;
      #pragma unroll
      for (int e = 0; e < 8; ++e) s += acc[mt][nt][e] * acc[mt][nt][e];
      atomicAdd(&vacc[col], s);
    }
}

// ---------------------------------------------------------------------------
// Blocked Cholesky (NB=32), f32, in place on Km
// ---------------------------------------------------------------------------
__global__ void k_chol_panel(float* __restrict__ Lm, int s) {
  __shared__ float T[32][33];
  __shared__ float idg[32];
  const int done = s * 32;
  const int tid = threadIdx.x;             // 1024 threads
  const int r = tid >> 5, c = tid & 31;
  T[r][c] = Lm[(size_t)(done + r) * Pp + done + c];
  __syncthreads();
  for (int j = 0; j < 32; ++j) {
    if (tid == 0) { T[j][j] = sqrtf(T[j][j]); idg[j] = 1.f / T[j][j]; }
    __syncthreads();
    if (r > j && c == j) T[r][j] *= idg[j];
    __syncthreads();
    if (r > j && c > j && c <= r) T[r][c] -= T[r][j] * T[c][j];
    __syncthreads();
  }
  Lm[(size_t)(done + r) * Pp + done + c] = T[r][c];
  __syncthreads();
  const int rem = Pp - done - 32;          // panel TRSM: one row per thread
  if (tid < rem) {
    const int i = done + 32 + tid;
    float a[32];
    #pragma unroll
    for (int j = 0; j < 32; ++j) a[j] = Lm[(size_t)i * Pp + done + j];
    #pragma unroll
    for (int j = 0; j < 32; ++j) {
      float v = a[j];
      #pragma unroll
      for (int k = 0; k < j; ++k) v -= a[k] * T[j][k];
      a[j] = v * idg[j];
    }
    #pragma unroll
    for (int j = 0; j < 32; ++j) Lm[(size_t)i * Pp + done + j] = a[j];
  }
}

__global__ void k_chol_trail(float* __restrict__ Lm, int s) {
  const int done = s * 32, base = done + 32;
  const int bi = blockIdx.x, bj = blockIdx.y;
  if (bj > bi) return;
  __shared__ float Pi[32][33], Pj[32][33];
  const int tid = threadIdx.x;             // 256
  #pragma unroll
  for (int q = 0; q < 4; ++q) {
    const int idx = tid + q * 256, rr = idx >> 5, cc = idx & 31;
    Pi[rr][cc] = Lm[(size_t)(base + bi * 32 + rr) * Pp + done + cc];
    Pj[rr][cc] = Lm[(size_t)(base + bj * 32 + rr) * Pp + done + cc];
  }
  __syncthreads();
  const int r = tid >> 3, cg = tid & 7;
  float a4[4] = {0.f, 0.f, 0.f, 0.f};
  #pragma unroll
  for (int k = 0; k < 32; ++k) {
    const float a = Pi[r][k];
    #pragma unroll
    for (int q = 0; q < 4; ++q) a4[q] += a * Pj[cg + q * 8][k];
  }
  const size_t rowg = (size_t)(base + bi * 32 + r) * Pp + base + bj * 32;
  #pragma unroll
  for (int q = 0; q < 4; ++q) Lm[rowg + cg + q * 8] -= a4[q];
}

// ---------------------------------------------------------------------------
// Blocked lower-triangular inverse: W = inv(L)
// ---------------------------------------------------------------------------
__global__ void k_trinv_diag(const float* __restrict__ Lm, float* __restrict__ W) {
  const int b = blockIdx.x, t = threadIdx.x;   // 32 threads, thread = column
  __shared__ float Ls[32][33];
  __shared__ float idg[32];
  for (int r = 0; r < 32; ++r) Ls[r][t] = Lm[(size_t)(b * 32 + r) * Pp + b * 32 + t];
  idg[t] = 1.f / Ls[t][t];
  __syncthreads();
  float x[32];
  #pragma unroll
  for (int r = 0; r < 32; ++r) {               // solve L x = e_t (x[r]=0 for r<t)
    float v = (r == t) ? 1.f : 0.f;
    #pragma unroll
    for (int k = 0; k < r; ++k) v -= Ls[r][k] * x[k];
    x[r] = v * idg[r];
  }
  for (int r = 0; r < 32; ++r) W[(size_t)(b * 32 + r) * Pp + b * 32 + t] = x[r];
}

__global__ void k_trinv_step(const float* __restrict__ Lm, float* __restrict__ W, int d) {
  const int j = blockIdx.x, i = j + d;         // W(i,j) = -W(i,i) * sum_k L(i,k) W(k,j)
  __shared__ float Sa[32][33], Sb[32][33], S[32][33];
  const int tid = threadIdx.x;                 // 256
  const int r = tid >> 3, cg = tid & 7;
  float a4[4] = {0.f, 0.f, 0.f, 0.f};
  for (int k = j; k < i; ++k) {
    #pragma unroll
    for (int q = 0; q < 4; ++q) {
      const int idx = tid + q * 256, rr = idx >> 5, cc = idx & 31;
      Sa[rr][cc] = Lm[(size_t)(i * 32 + rr) * Pp + k * 32 + cc];
      Sb[rr][cc] = W [(size_t)(k * 32 + rr) * Pp + j * 32 + cc];
    }
    __syncthreads();
    #pragma unroll
    for (int kk = 0; kk < 32; ++kk) {
      const float a = Sa[r][kk];
      #pragma unroll
      for (int q = 0; q < 4; ++q) a4[q] += a * Sb[kk][cg + q * 8];
    }
    __syncthreads();
  }
  #pragma unroll
  for (int q = 0; q < 4; ++q) S[r][cg + q * 8] = a4[q];
  #pragma unroll
  for (int q = 0; q < 4; ++q) {
    const int idx = tid + q * 256, rr = idx >> 5, cc = idx & 31;
    Sa[rr][cc] = W[(size_t)(i * 32 + rr) * Pp + i * 32 + cc];
  }
  __syncthreads();
  float o4[4] = {0.f, 0.f, 0.f, 0.f};
  #pragma unroll
  for (int kk = 0; kk < 32; ++kk) {
    const float a = Sa[r][kk];
    #pragma unroll
    for (int q = 0; q < 4; ++q) o4[q] += a * S[kk][cg + q * 8];
  }
  #pragma unroll
  for (int q = 0; q < 4; ++q)
    W[(size_t)(i * 32 + r) * Pp + j * 32 + cg + q * 8] = -o4[q];
}

__global__ void k_f32_to_bf16(const float* __restrict__ W, __bf16* __restrict__ Wh) {
  const size_t i = (size_t)blockIdx.x * 256 + threadIdx.x;
  Wh[i] = (__bf16)W[i];
}

__global__ void k_var_final(const float* __restrict__ vacc, const bool* __restrict__ msk,
                            float* __restrict__ out) {
  const int i = blockIdx.x * 256 + threadIdx.x;
  out[i] = msk[i] ? (1.f - vacc[i]) : 0.f;
}

// ---------------------------------------------------------------------------
extern "C" void kernel_launch(void* const* d_in, const int* in_sizes, int n_in,
                              void* d_out, int out_size, void* d_ws, size_t ws_size,
                              hipStream_t stream) {
  const float* x     = (const float*)d_in[0];
  const bool*  msk   = (const bool*)d_in[1];
  const float* Z     = (const float*)d_in[2];
  const float* gamma = (const float*)d_in[4];   // targets (d_in[3]) unused in outputs
  const float* sigma = (const float*)d_in[5];
  float* out = (float*)d_out;
  char*  ws  = (char*)d_ws;

  float*  st   = (float*)(ws + OFF_STATS);
  float*  prm  = (float*)(ws + OFF_PARAMS);
  float*  zn   = (float*)(ws + OFF_ZN);
  float*  xn   = (float*)(ws + OFF_XN);
  __bf16* zg   = (__bf16*)(ws + OFF_ZG);
  __bf16* xg   = (__bf16*)(ws + OFF_XG);
  float*  Km   = (float*)(ws + OFF_K);
  float*  W    = (float*)(ws + OFF_LINV);
  __bf16* Wh   = (__bf16*)(ws + OFF_LINVH);
  __bf16* Kx   = (__bf16*)(ws + OFF_KX);
  float*  vacc = (float*)(ws + OFF_VACC);

  hipMemsetAsync(st, 0, 4096, stream);
  hipMemsetAsync(W, 0, (size_t)Pp * Pp * 4, stream);  // upper triangle stays 0
  hipMemsetAsync(vacc, 0, (size_t)Nn * 4, stream);

  k_stats   <<<256, 256, 0, stream>>>(x, msk, st);
  k_finalize<<<1,   256, 0, stream>>>(st, gamma, sigma, prm);
  k_prep_z  <<<Pp,  256, 0, stream>>>(Z, prm, zg, zn);
  k_prep_x  <<<Nn,  256, 0, stream>>>(x, prm, xg, xn);

  k_gemm_zz<<<dim3(Pp / 128, Pp / 128), 256, 0, stream>>>(zg, zn, prm, Km);

  for (int s = 0; s < Pp / 32; ++s) {
    k_chol_panel<<<1, 1024, 0, stream>>>(Km, s);
    const int ntr = Pp / 32 - 1 - s;
    if (ntr > 0) k_chol_trail<<<dim3(ntr, ntr), 256, 0, stream>>>(Km, s);
  }

  k_trinv_diag<<<Pp / 32, 32, 0, stream>>>(Km, W);
  for (int d = 1; d < Pp / 32; ++d)
    k_trinv_step<<<Pp / 32 - d, 256, 0, stream>>>(Km, W, d);
  k_f32_to_bf16<<<(Pp * Pp) / 256, 256, 0, stream>>>(W, Wh);

  k_gemm_xz<<<dim3(Nn / 128, Pp / 128), 256, 0, stream>>>(xg, zg, xn, zn, msk, Kx, out);
  k_gemm_v <<<dim3(Pp / 128, Nn / 128), 256, 0, stream>>>(Wh, Kx, vacc);
  k_var_final<<<Nn / 256, 256, 0, stream>>>(vacc, msk, out + (size_t)Nn * Pp);
}
